// PerceiverAR_Fixed_Token_Per_Latent_Scaling_25615184953792
// MI455X (gfx1250) — matrix operand order repaired and checked by reading, not compiled, not run on hardware
//
#include <hip/hip_runtime.h>
#include <hip/hip_bf16.h>
#include <cstdint>

// ---------------------------------------------------------------------------
// Model constants (from the reference)
// ---------------------------------------------------------------------------
#define S_  2048
#define D_  1024
#define H_  16
#define L_  256
#define TPL_ 8
#define NL_ 6
#define HD_ 64
#define DFF_ 4096

typedef __attribute__((ext_vector_type(16))) __bf16 v16bf;
typedef __attribute__((ext_vector_type(8)))  float  v8f;
typedef int v4i_ __attribute__((__vector_size__(16)));

union FragU {
    unsigned u[8];
    v16bf    v;
};

#define AS1 __attribute__((address_space(1)))
#define AS3 __attribute__((address_space(3)))

// CDNA5 async global->LDS staging (guarded; fallback = plain 16B copy)
#if defined(__gfx1250__) && __has_builtin(__builtin_amdgcn_global_load_async_to_lds_b128) && __has_builtin(__builtin_amdgcn_s_wait_asynccnt)
#define USE_ASYNC_LDS 1
#else
#define USE_ASYNC_LDS 0
#endif

__device__ __forceinline__ void copy16_g2l(const __bf16* g, __bf16* l) {
#if USE_ASYNC_LDS
    __builtin_amdgcn_global_load_async_to_lds_b128(
        (AS1 v4i_*)(uintptr_t)g,
        (AS3 v4i_*)(unsigned)(uintptr_t)l,
        0, 0);
#else
    *(uint4*)l = *(const uint4*)g;
#endif
}

__device__ __forceinline__ void async_join() {
#if USE_ASYNC_LDS
    __builtin_amdgcn_s_wait_asynccnt(0);
#endif
}

// ---------------------------------------------------------------------------
// Elementwise helpers
// ---------------------------------------------------------------------------
__global__ void add2_kernel(float* __restrict__ out, const float* __restrict__ a,
                            const float* __restrict__ b, int n) {
    int i = blockIdx.x * 256 + threadIdx.x;
    if (i < n) out[i] = a[i] + b[i];
}

__global__ void f32_to_bf16_kernel(const float* __restrict__ in, __bf16* __restrict__ out,
                                   long long n) {
    long long i = ((long long)blockIdx.x * 256 + threadIdx.x) * 4;
    if (i + 4 <= n) {
        float4 f = *(const float4*)(in + i);
        out[i + 0] = (__bf16)f.x;
        out[i + 1] = (__bf16)f.y;
        out[i + 2] = (__bf16)f.z;
        out[i + 3] = (__bf16)f.w;
    }
}

__global__ void zero_bf16_kernel(__bf16* __restrict__ p, int n) {
    int i = blockIdx.x * 64 + threadIdx.x;
    if (i < n) p[i] = (__bf16)0.f;
}

// 32x32 LDS-tiled bf16 transpose: out[c][r] = in[r][c]
__global__ __launch_bounds__(256)
void transpose_bf16_kernel(const __bf16* __restrict__ in, __bf16* __restrict__ out,
                           int rows, int cols) {
    __shared__ __bf16 t[32][33];
    int bx = blockIdx.x * 32;   // col base
    int by = blockIdx.y * 32;   // row base
    int tx = threadIdx.x & 31, ty = threadIdx.x >> 5;   // 8 row-groups
    #pragma unroll
    for (int i = ty; i < 32; i += 8)
        t[i][tx] = in[(long long)(by + i) * cols + bx + tx];
    __syncthreads();
    #pragma unroll
    for (int i = ty; i < 32; i += 8)
        out[(long long)(bx + i) * rows + by + tx] = t[tx][i];
}

// grouped V sums for adj_up path: vgs[j][d] = sum_{t=0..7} v[8j+t][d]
__global__ void group_vsum_kernel(const float* __restrict__ v, float* __restrict__ vgs, int n) {
    int i = blockIdx.x * 256 + threadIdx.x;
    if (i >= n) return;
    int j = i >> 10;        // / D_
    int d = i & (D_ - 1);
    float s = 0.f;
    #pragma unroll
    for (int t = 0; t < 8; ++t) s += v[(long long)(8 * j + t) * D_ + d];
    vgs[i] = s;
}

// ---------------------------------------------------------------------------
// LayerNorm: one block per row; optional fused "add"; f32 and/or bf16 outputs
// ---------------------------------------------------------------------------
__global__ __launch_bounds__(256)
void layernorm_rows(const float* __restrict__ in, const float* __restrict__ add,
                    const float* __restrict__ g, const float* __restrict__ bta,
                    float* __restrict__ outF, __bf16* __restrict__ outB, int Dd) {
    __shared__ float red[256];
    long long row = blockIdx.x;
    const float* ip = in + row * Dd;
    const float* ap = add ? add + row * Dd : nullptr;
    float s = 0.f;
    for (int d = threadIdx.x; d < Dd; d += 256) {
        float v = ip[d] + (ap ? ap[d] : 0.f);
        s += v;
    }
    red[threadIdx.x] = s; __syncthreads();
    for (int o = 128; o > 0; o >>= 1) {
        if (threadIdx.x < o) red[threadIdx.x] += red[threadIdx.x + o];
        __syncthreads();
    }
    float mean = red[0] / (float)Dd; __syncthreads();
    float vs = 0.f;
    for (int d = threadIdx.x; d < Dd; d += 256) {
        float v = ip[d] + (ap ? ap[d] : 0.f) - mean;
        vs += v * v;
    }
    red[threadIdx.x] = vs; __syncthreads();
    for (int o = 128; o > 0; o >>= 1) {
        if (threadIdx.x < o) red[threadIdx.x] += red[threadIdx.x + o];
        __syncthreads();
    }
    float rstd = rsqrtf(red[0] / (float)Dd + 1e-5f);
    for (int d = threadIdx.x; d < Dd; d += 256) {
        float v = ip[d] + (ap ? ap[d] : 0.f);
        float r = (v - mean) * rstd * g[d] + bta[d];
        if (outF) outF[row * Dd + d] = r;
        if (outB) outB[row * Dd + d] = (__bf16)r;
    }
}

// ---------------------------------------------------------------------------
// Masked softmax (in-place f32 + bf16 mirror). Row i valid for j < (i+1)*mult.
// ---------------------------------------------------------------------------
__global__ __launch_bounds__(256)
void softmax_rows(float* __restrict__ sc, __bf16* __restrict__ scb,
                  int rowlen, int rowsPer, int mult) {
    __shared__ float red[256];
    long long row = blockIdx.x;
    int i = (int)(row % rowsPer);
    int limit = (i + 1) * mult;
    if (limit > rowlen) limit = rowlen;
    float* p = sc + row * (long long)rowlen;
    __bf16* pb = scb + row * (long long)rowlen;
    float mx = -1e30f;
    for (int j = threadIdx.x; j < limit; j += 256) mx = fmaxf(mx, p[j]);
    red[threadIdx.x] = mx; __syncthreads();
    for (int o = 128; o > 0; o >>= 1) {
        if (threadIdx.x < o) red[threadIdx.x] = fmaxf(red[threadIdx.x], red[threadIdx.x + o]);
        __syncthreads();
    }
    mx = red[0]; __syncthreads();
    float s = 0.f;
    for (int j = threadIdx.x; j < limit; j += 256) {
        float e = __expf(p[j] - mx);
        p[j] = e; s += e;
    }
    red[threadIdx.x] = s; __syncthreads();
    for (int o = 128; o > 0; o >>= 1) {
        if (threadIdx.x < o) red[threadIdx.x] += red[threadIdx.x + o];
        __syncthreads();
    }
    float inv = 1.f / (red[0] + 1e-30f);
    for (int j = threadIdx.x; j < rowlen; j += 256) {
        float v = (j < limit) ? p[j] * inv : 0.f;
        p[j] = v;
        pb[j] = (__bf16)v;
    }
}

// ---------------------------------------------------------------------------
// Batched WMMA GEMM, bf16 in / f32 accumulate:
//   C[b] = alpha * A[b] x B[b](^T) (+bias)(gelu)(+res)
// Tile 128x64x32, 256 threads (8 waves), wave = 2 M-subtiles x 2 N-subtiles.
// LDS staged via async global->LDS b128 (raw bf16 copy, no converts in loop).
// Dims must satisfy 128|M, 64|N, 32|K (true for every call site here).
// ---------------------------------------------------------------------------
#define TM 128
#define TN 64
#define TK 32

template <bool TRANSB>
__global__ __launch_bounds__(256)
void wmma_gemm(const __bf16* __restrict__ A, int lda, long long sA,
               const __bf16* __restrict__ B, int ldb, long long sB,
               float* __restrict__ Cf, __bf16* __restrict__ Cb,
               int ldc, long long sC,
               const float* __restrict__ bias,
               const float* __restrict__ res, int ldr, long long sR,
               int M, int N, int K, float alpha, int gelu) {
    __shared__ __bf16 As[TM * TK];          // [m][k]
    __shared__ __bf16 Bs[TN * TK];          // NN: [k][TN] ; NT: [n][TK]

    long long bz = blockIdx.z;
    A += bz * sA;
    B += bz * sB;
    if (Cf) Cf += bz * sC;
    if (Cb) Cb += bz * sC;
    if (res) res += bz * sR;

    const int m0 = blockIdx.y * TM;
    const int n0 = blockIdx.x * TN;
    const int tid  = threadIdx.x;
    const int wave = tid >> 5;
    const int lane = tid & 31;
    const int lh   = lane >> 4;
    const int lm   = lane & 15;
    const int mSub0 = (wave >> 1) * 2;      // wave covers mSub0, mSub0+1 (of 8)
    const int nSub0 = (wave & 1) * 2;       // wave covers nSub0, nSub0+1 (of 4)

    v8f acc00 = {}, acc01 = {}, acc10 = {}, acc11 = {};

    for (int k0 = 0; k0 < K; k0 += TK) {
        // ---- stage A (128x32 bf16, raw copy) ----
        #pragma unroll
        for (int i = 0; i < 2; ++i) {
            int c  = tid + i * 256;          // 512 chunks of 8 bf16
            int mm = c >> 2, ck = (c & 3) * 8;
            copy16_g2l(A + (long long)(m0 + mm) * lda + k0 + ck, &As[mm * TK + ck]);
        }
        // ---- stage B (2048 bf16 = 256 chunks) ----
        {
            int c = tid;
            if (TRANSB) {
                int nn = c >> 2, ck = (c & 3) * 8;
                copy16_g2l(B + (long long)(n0 + nn) * ldb + k0 + ck, &Bs[nn * TK + ck]);
            } else {
                int kk = c >> 3, cn = (c & 7) * 8;
                copy16_g2l(B + (long long)(k0 + kk) * ldb + n0 + cn, &Bs[kk * TN + cn]);
            }
        }
        async_join();
        __syncthreads();

        // ---- fragments (contiguous k = 16*lh + s, same order in A and B) ----
        FragU fa0, fa1, fb0, fb1;
        #pragma unroll
        for (int r = 0; r < 8; ++r) {
            int kk = lh * 16 + 2 * r;
            fa0.u[r] = *(const unsigned*)&As[(mSub0 * 16 + lm) * TK + kk];
            fa1.u[r] = *(const unsigned*)&As[((mSub0 + 1) * 16 + lm) * TK + kk];
        }
        if (TRANSB) {
            #pragma unroll
            for (int r = 0; r < 8; ++r) {
                int kk = lh * 16 + 2 * r;
                fb0.u[r] = *(const unsigned*)&Bs[(nSub0 * 16 + lm) * TK + kk];
                fb1.u[r] = *(const unsigned*)&Bs[((nSub0 + 1) * 16 + lm) * TK + kk];
            }
        } else {
            #pragma unroll
            for (int r = 0; r < 8; ++r) {
                int kk = lh * 16 + 2 * r;
                fb0.v[2 * r]     = Bs[kk * TN + nSub0 * 16 + lm];
                fb0.v[2 * r + 1] = Bs[(kk + 1) * TN + nSub0 * 16 + lm];
                fb1.v[2 * r]     = Bs[kk * TN + (nSub0 + 1) * 16 + lm];
                fb1.v[2 * r + 1] = Bs[(kk + 1) * TN + (nSub0 + 1) * 16 + lm];
            }
        }
        acc00 = __builtin_amdgcn_wmma_f32_16x16x32_bf16(false, fa0.v, false, fb0.v, (short)0, acc00, false, false);
        acc01 = __builtin_amdgcn_wmma_f32_16x16x32_bf16(false, fa0.v, false, fb1.v, (short)0, acc01, false, false);
        acc10 = __builtin_amdgcn_wmma_f32_16x16x32_bf16(false, fa1.v, false, fb0.v, (short)0, acc10, false, false);
        acc11 = __builtin_amdgcn_wmma_f32_16x16x32_bf16(false, fa1.v, false, fb1.v, (short)0, acc11, false, false);
        __syncthreads();
    }

    // ---- epilogue ----
    #pragma unroll
    for (int j = 0; j < 2; ++j) {
        int gmBase = m0 + (mSub0 + j) * 16 + lh * 8;
        #pragma unroll
        for (int t = 0; t < 2; ++t) {
            int gn = n0 + (nSub0 + t) * 16 + lm;
            v8f a = (j == 0) ? (t == 0 ? acc00 : acc01) : (t == 0 ? acc10 : acc11);
            float bv = bias ? bias[gn] : 0.f;
            #pragma unroll
            for (int r = 0; r < 8; ++r) {
                int gm = gmBase + r;
                float v = a[r] * alpha + bv;
                if (gelu) v = 0.5f * v * (1.f + erff(v * 0.70710678118654752f));
                if (res)  v += res[(long long)gm * ldr + gn];
                long long o = (long long)gm * ldc + gn;
                if (Cf) Cf[o] = v;
                if (Cb) Cb[o] = (__bf16)v;
            }
        }
    }
}

// ---------------------------------------------------------------------------
// Decoder flash attention (per head, 16-query tile per wave/block) with the
// supplied-adjacency mix path:
//   out = (mix * softmaxO + (1-mix) * adjV_unmasked)
//         / (mix + (1-mix) * 8*rowsum(adj[qg]) + 1e-9)
// bf16 q/k fragments straight from global; V fragments from a transposed
// bf16 copy vT[D][S] (64-elem zeroed tail pad) so every k-pair is an aligned
// dword and r-groups merge into global_load_b128 (affine addresses, no clamp).
// ---------------------------------------------------------------------------
__global__ __launch_bounds__(32)
void dec_flash_attn(const __bf16* __restrict__ q, const __bf16* __restrict__ k,
                    const __bf16* __restrict__ vT, const float* __restrict__ vf,
                    const float* __restrict__ adj, const float* __restrict__ vgs,
                    const float* __restrict__ mixp, __bf16* __restrict__ out) {
    __shared__ float  sS[16 * 16];
    __shared__ __bf16 sP[16 * 16];
    __shared__ float  sScale[16];
    __shared__ float  sM[16];
    __shared__ float  sZ[16];
    __shared__ float  sO[16 * 64];
    __shared__ float  sAdjV[16 * 64];

    int qt = blockIdx.x;
    int h  = blockIdx.y;
    int lane = threadIdx.x;
    int lh = lane >> 4, lm = lane & 15;
    float mixv = mixp[0];

    // Q fragments (16x64 -> two K=32 A-fragments), dword loads of bf16 pairs
    FragU a0, a1;
    {
        const __bf16* qp = q + (long long)(qt * 16 + lm) * D_ + h * HD_;
        #pragma unroll
        for (int r = 0; r < 8; ++r) {
            int kk = lh * 16 + 2 * r;
            a0.u[r] = *(const unsigned*)(qp + kk);
            a1.u[r] = *(const unsigned*)(qp + kk + 32);
        }
    }

    v8f o0 = {}, o1 = {}, o2 = {}, o3 = {};
    if (lane < 16) { sM[lane] = -1e30f; sZ[lane] = 0.f; }
    __syncthreads();

    for (int kt = 0; kt <= qt; ++kt) {
        FragU b0, b1;
        {
            const __bf16* kp = k + (long long)(kt * 16 + lm) * D_ + h * HD_;
            #pragma unroll
            for (int r = 0; r < 8; ++r) {
                int f = lh * 16 + 2 * r;
                b0.u[r] = *(const unsigned*)(kp + f);
                b1.u[r] = *(const unsigned*)(kp + f + 32);
            }
        }
        v8f sacc = {};
        sacc = __builtin_amdgcn_wmma_f32_16x16x32_bf16(false, a0.v, false, b0.v, (short)0, sacc, false, false);
        sacc = __builtin_amdgcn_wmma_f32_16x16x32_bf16(false, a1.v, false, b1.v, (short)0, sacc, false, false);

        #pragma unroll
        for (int r = 0; r < 8; ++r) {
            int qr = qt * 16 + lh * 8 + r;
            int kc = kt * 16 + lm;
            float val = sacc[r] * 0.125f;       // HD^-0.5
            if (kc > qr) val = -1e30f;
            sS[(lh * 8 + r) * 16 + lm] = val;
        }
        __syncthreads();

        if (lane < 16) {
            float m_old = sM[lane];
            float tmax = m_old;
            #pragma unroll
            for (int j = 0; j < 16; ++j) tmax = fmaxf(tmax, sS[lane * 16 + j]);
            float sc = __expf(m_old - tmax);
            float zs = 0.f;
            #pragma unroll
            for (int j = 0; j < 16; ++j) {
                float p = __expf(sS[lane * 16 + j] - tmax);
                sP[lane * 16 + j] = (__bf16)p;
                zs += p;
            }
            sZ[lane] = sZ[lane] * sc + zs;
            sM[lane] = tmax;
            sScale[lane] = sc;
        }
        __syncthreads();

        #pragma unroll
        for (int r = 0; r < 8; ++r) {
            float sc = sScale[lh * 8 + r];
            o0[r] *= sc; o1[r] *= sc; o2[r] *= sc; o3[r] *= sc;
        }

        // P fragment (K padded 16 -> 32 with zeros); pairs are dword LDS reads
        FragU pf;
        #pragma unroll
        for (int r = 0; r < 8; ++r) {
            int kk = lh * 16 + 2 * r;
            pf.u[r] = (kk < 16) ? *(const unsigned*)&sP[lm * 16 + kk] : 0u;
        }
        // V fragments from vT[D][S]: affine dword loads (merge into b128).
        // Over-reads past S on the last tile land in the zeroed tail pad and
        // are multiplied by the zero half of the P fragment.
        #pragma unroll
        for (int t = 0; t < 4; ++t) {
            const __bf16* vp = vT + (long long)(h * HD_ + t * 16 + lm) * S_ + kt * 16 + lh * 16;
            FragU vfr;
            #pragma unroll
            for (int r = 0; r < 8; ++r) {
                vfr.u[r] = *(const unsigned*)(vp + 2 * r);
            }
            if (t == 0) o0 = __builtin_amdgcn_wmma_f32_16x16x32_bf16(false, pf.v, false, vfr.v, (short)0, o0, false, false);
            if (t == 1) o1 = __builtin_amdgcn_wmma_f32_16x16x32_bf16(false, pf.v, false, vfr.v, (short)0, o1, false, false);
            if (t == 2) o2 = __builtin_amdgcn_wmma_f32_16x16x32_bf16(false, pf.v, false, vfr.v, (short)0, o2, false, false);
            if (t == 3) o3 = __builtin_amdgcn_wmma_f32_16x16x32_bf16(false, pf.v, false, vfr.v, (short)0, o3, false, false);
        }
        __syncthreads();
    }

    #pragma unroll
    for (int r = 0; r < 8; ++r) {
        int row = lh * 8 + r;
        float invZ = 1.f / (sZ[row] + 1e-30f);
        sO[row * 64 +  0 + lm] = o0[r] * invZ;
        sO[row * 64 + 16 + lm] = o1[r] * invZ;
        sO[row * 64 + 32 + lm] = o2[r] * invZ;
        sO[row * 64 + 48 + lm] = o3[r] * invZ;
    }
    __syncthreads();

    // adjacency path (nearest-neighbor upscaled latent attention)
    if (lane < 16) {
        int qidx = qt * 16 + lane;
        int qg = qidx >> 3;
        const float* adjrow = adj + ((long long)h * L_ + qg) * (long long)L_;
        float* av = &sAdjV[lane * 64];
        for (int d = 0; d < 64; ++d) av[d] = 0.f;
        float asum = 0.f;
        for (int j = 0; j <= qg; ++j) {
            float w = adjrow[j];
            asum += 8.f * w;                       // full-row sum (pre-mask renorm)
            if (j < qg) {
                const float* vg = vgs + (long long)j * D_ + h * HD_;
                for (int d = 0; d < 64; ++d) av[d] += w * vg[d];
            } else {
                int cnt = qidx - 8 * qg + 1;       // partial group up to q
                for (int t = 0; t < cnt; ++t) {
                    const float* vr = vf + (long long)(8 * qg + t) * D_ + h * HD_;
                    for (int d = 0; d < 64; ++d) av[d] += w * vr[d];
                }
            }
        }
        sM[lane] = mixv + (1.f - mixv) * asum + 1e-9f;   // reuse sM as denom
    }
    __syncthreads();

    for (int idx = lane; idx < 16 * 64; idx += 32) {
        int row = idx >> 6, d = idx & 63;
        float val = (mixv * sO[idx] + (1.f - mixv) * sAdjV[idx]) / sM[row];
        out[(long long)(qt * 16 + row) * D_ + h * HD_ + d] = (__bf16)val;
    }
}

// ---------------------------------------------------------------------------
// Host orchestration
// ---------------------------------------------------------------------------
extern "C" void kernel_launch(void* const* d_in, const int* in_sizes, int n_in,
                              void* d_out, int out_size, void* d_ws, size_t ws_size,
                              hipStream_t stream) {
    const float* hidden = (const float*)d_in[0];
    const float* ipos   = (const float*)d_in[1];
    const float* lpos   = (const float*)d_in[2];
    const float* larr   = (const float*)d_in[3];
    const float* decq   = (const float*)d_in[4];
    const float* eWq = (const float*)d_in[5];
    const float* eWk = (const float*)d_in[6];
    const float* eWv = (const float*)d_in[7];
    const float* eWo = (const float*)d_in[8];
    const float* ebo = (const float*)d_in[9];
    const float* lWq = (const float*)d_in[10];
    const float* lWk = (const float*)d_in[11];
    const float* lWv = (const float*)d_in[12];
    const float* lWo = (const float*)d_in[13];
    const float* lbo = (const float*)d_in[14];
    const float* ln1g = (const float*)d_in[15];
    const float* ln1b = (const float*)d_in[16];
    const float* ln2g = (const float*)d_in[17];
    const float* ln2b = (const float*)d_in[18];
    const float* lW1 = (const float*)d_in[19];
    const float* lb1 = (const float*)d_in[20];
    const float* lW2 = (const float*)d_in[21];
    const float* lb2 = (const float*)d_in[22];
    const float* dWq = (const float*)d_in[23];
    const float* dWk = (const float*)d_in[24];
    const float* dWv = (const float*)d_in[25];
    const float* dWo = (const float*)d_in[26];
    const float* dbo = (const float*)d_in[27];
    const float* n1g = (const float*)d_in[28];
    const float* n1b = (const float*)d_in[29];
    const float* n2g = (const float*)d_in[30];
    const float* n2b = (const float*)d_in[31];
    const float* fing = (const float*)d_in[32];
    const float* finb = (const float*)d_in[33];
    const float* mixp = (const float*)d_in[34];

    // workspace allocator
    size_t off = 0;
    auto allocF = [&](size_t n) {
        float* p = (float*)((char*)d_ws + off);
        off = (off + n * 4 + 255) & ~(size_t)255;
        return p;
    };
    auto allocB = [&](size_t n) {
        __bf16* p = (__bf16*)((char*)d_ws + off);
        off = (off + n * 2 + 255) & ~(size_t)255;
        return p;
    };

    // f32 masters
    float* xbuf   = allocF((size_t)S_ * D_);
    float* latbuf = allocF((size_t)L_ * D_);
    float* encsc  = allocF((size_t)H_ * L_ * S_);
    float* latsc  = allocF((size_t)H_ * L_ * L_);   // adj after last layer
    float* dvv    = allocF((size_t)S_ * D_);
    float* vgsb   = allocF((size_t)L_ * D_);
    float* ybuf   = allocF((size_t)S_ * D_);

    // bf16 weight copies
    __bf16* eWq_b = allocB((size_t)D_ * D_);
    __bf16* eWk_b = allocB((size_t)D_ * D_);
    __bf16* eWv_b = allocB((size_t)D_ * D_);
    __bf16* eWo_b = allocB((size_t)D_ * D_);
    __bf16* lWq_b = allocB((size_t)NL_ * D_ * D_);
    __bf16* lWk_b = allocB((size_t)NL_ * D_ * D_);
    __bf16* lWv_b = allocB((size_t)NL_ * D_ * D_);
    __bf16* lWo_b = allocB((size_t)NL_ * D_ * D_);
    __bf16* lW1_b = allocB((size_t)NL_ * D_ * DFF_);
    __bf16* lW2_b = allocB((size_t)NL_ * DFF_ * D_);
    __bf16* dWq_b = allocB((size_t)D_ * D_);
    __bf16* dWk_b = allocB((size_t)D_ * D_);
    __bf16* dWv_b = allocB((size_t)D_ * D_);
    __bf16* dWo_b = allocB((size_t)D_ * D_);

    // bf16 activations
    __bf16* xn_b    = allocB((size_t)S_ * D_);
    __bf16* hbuf_b  = allocB((size_t)L_ * D_);
    __bf16* qlat_b  = allocB((size_t)L_ * D_);
    __bf16* kx_b    = allocB((size_t)S_ * D_);
    __bf16* vx_b    = allocB((size_t)S_ * D_);
    __bf16* encsc_b = allocB((size_t)H_ * L_ * S_);
    __bf16* ao_b    = allocB((size_t)L_ * D_);
    __bf16* ff_b    = allocB((size_t)L_ * DFF_);
    __bf16* latsc_b = allocB((size_t)H_ * L_ * L_);
    __bf16* ql_b    = allocB((size_t)L_ * D_);
    __bf16* kl_b    = allocB((size_t)L_ * D_);
    __bf16* vl_b    = allocB((size_t)L_ * D_);
    __bf16* dqin_b  = allocB((size_t)S_ * D_);
    __bf16* knb_b   = allocB((size_t)S_ * D_);
    __bf16* dq_b    = allocB((size_t)S_ * D_);
    __bf16* dk_b    = allocB((size_t)S_ * D_);
    __bf16* dv_b    = allocB((size_t)S_ * D_);
    __bf16* dvT_b   = allocB((size_t)D_ * S_ + 64);   // +64 zeroed tail pad
    __bf16* dao_b   = allocB((size_t)S_ * D_);

    auto cvt = [&](const float* src, __bf16* dst, long long n) {
        f32_to_bf16_kernel<<<(unsigned)((n / 4 + 255) / 256), 256, 0, stream>>>(src, dst, n);
    };
    auto gemmNN = [&](const __bf16* A, int lda, long long sA,
                      const __bf16* Bm, int ldb, long long sB,
                      float* Cf, __bf16* Cb, int ldc, long long sC,
                      const float* bias, const float* res, int ldr, long long sR,
                      int M, int N, int K, float alpha, int gelu, int batch) {
        dim3 g(N / TN, M / TM, batch);
        wmma_gemm<false><<<g, 256, 0, stream>>>(A, lda, sA, Bm, ldb, sB, Cf, Cb, ldc, sC,
                                                bias, res, ldr, sR, M, N, K, alpha, gelu);
    };
    auto gemmNT = [&](const __bf16* A, int lda, long long sA,
                      const __bf16* Bm, int ldb, long long sB,
                      float* Cf, __bf16* Cb, int ldc, long long sC,
                      int M, int N, int K, float alpha, int batch) {
        dim3 g(N / TN, M / TM, batch);
        wmma_gemm<true><<<g, 256, 0, stream>>>(A, lda, sA, Bm, ldb, sB, Cf, Cb, ldc, sC,
                                               nullptr, nullptr, 0, 0, M, N, K, alpha, 0);
    };

    // ---- one-time bf16 weight conversion ----
    cvt(eWq, eWq_b, (long long)D_ * D_);
    cvt(eWk, eWk_b, (long long)D_ * D_);
    cvt(eWv, eWv_b, (long long)D_ * D_);
    cvt(eWo, eWo_b, (long long)D_ * D_);
    cvt(lWq, lWq_b, (long long)NL_ * D_ * D_);
    cvt(lWk, lWk_b, (long long)NL_ * D_ * D_);
    cvt(lWv, lWv_b, (long long)NL_ * D_ * D_);
    cvt(lWo, lWo_b, (long long)NL_ * D_ * D_);
    cvt(lW1, lW1_b, (long long)NL_ * D_ * DFF_);
    cvt(lW2, lW2_b, (long long)NL_ * DFF_ * D_);
    cvt(dWq, dWq_b, (long long)D_ * D_);
    cvt(dWk, dWk_b, (long long)D_ * D_);
    cvt(dWv, dWv_b, (long long)D_ * D_);
    cvt(dWo, dWo_b, (long long)D_ * D_);

    // ---- init: x = hidden + pos; latents = latent_array + latent_pos ----
    add2_kernel<<<(S_ * D_ + 255) / 256, 256, 0, stream>>>(xbuf, hidden, ipos, S_ * D_);
    add2_kernel<<<(L_ * D_ + 255) / 256, 256, 0, stream>>>(latbuf, larr, lpos, L_ * D_);

    // ---- encoder cross attention ----
    layernorm_rows<<<S_, 256, 0, stream>>>(xbuf, nullptr, n1g, n1b, nullptr, xn_b, D_);
    layernorm_rows<<<L_, 256, 0, stream>>>(latbuf, nullptr, n1g, n1b, nullptr, hbuf_b, D_);
    gemmNN(hbuf_b, D_, 0, eWq_b, D_, 0, nullptr, qlat_b, D_, 0, nullptr, nullptr, 0, 0, L_, D_, D_, 1.f, 0, 1);
    gemmNN(xn_b,   D_, 0, eWk_b, D_, 0, nullptr, kx_b,   D_, 0, nullptr, nullptr, 0, 0, S_, D_, D_, 1.f, 0, 1);
    gemmNN(xn_b,   D_, 0, eWv_b, D_, 0, nullptr, vx_b,   D_, 0, nullptr, nullptr, 0, 0, S_, D_, D_, 1.f, 0, 1);
    gemmNT(qlat_b, D_, HD_, kx_b, D_, HD_, encsc, nullptr, S_, (long long)L_ * S_, L_, S_, HD_, 0.125f, H_);
    softmax_rows<<<H_ * L_, 256, 0, stream>>>(encsc, encsc_b, S_, L_, TPL_);
    gemmNN(encsc_b, S_, (long long)L_ * S_, vx_b, D_, HD_, nullptr, ao_b, D_, HD_,
           nullptr, nullptr, 0, 0, L_, HD_, S_, 1.f, 0, H_);
    gemmNN(ao_b, D_, 0, eWo_b, D_, 0, latbuf, nullptr, D_, 0, ebo, latbuf, D_, 0, L_, D_, D_, 1.f, 0, 1);

    // ---- latent self-attention stack ----
    for (int i = 0; i < NL_; ++i) {
        const size_t wofs = (size_t)i * D_ * D_;
        layernorm_rows<<<L_, 256, 0, stream>>>(latbuf, nullptr, ln1g + (size_t)i * D_, ln1b + (size_t)i * D_,
                                               nullptr, hbuf_b, D_);
        gemmNN(hbuf_b, D_, 0, lWq_b + wofs, D_, 0, nullptr, ql_b, D_, 0, nullptr, nullptr, 0, 0, L_, D_, D_, 1.f, 0, 1);
        gemmNN(hbuf_b, D_, 0, lWk_b + wofs, D_, 0, nullptr, kl_b, D_, 0, nullptr, nullptr, 0, 0, L_, D_, D_, 1.f, 0, 1);
        gemmNN(hbuf_b, D_, 0, lWv_b + wofs, D_, 0, nullptr, vl_b, D_, 0, nullptr, nullptr, 0, 0, L_, D_, D_, 1.f, 0, 1);
        gemmNT(ql_b, D_, HD_, kl_b, D_, HD_, latsc, nullptr, L_, (long long)L_ * L_, L_, L_, HD_, 0.125f, H_);
        softmax_rows<<<H_ * L_, 256, 0, stream>>>(latsc, latsc_b, L_, L_, 1);   // causal; last = adj
        gemmNN(latsc_b, L_, (long long)L_ * L_, vl_b, D_, HD_, nullptr, ao_b, D_, HD_,
               nullptr, nullptr, 0, 0, L_, HD_, L_, 1.f, 0, H_);
        gemmNN(ao_b, D_, 0, lWo_b + wofs, D_, 0, latbuf, nullptr, D_, 0,
               lbo + (size_t)i * D_, latbuf, D_, 0, L_, D_, D_, 1.f, 0, 1);
        layernorm_rows<<<L_, 256, 0, stream>>>(latbuf, nullptr, ln2g + (size_t)i * D_, ln2b + (size_t)i * D_,
                                               nullptr, hbuf_b, D_);
        gemmNN(hbuf_b, D_, 0, lW1_b + (size_t)i * D_ * DFF_, DFF_, 0, nullptr, ff_b, DFF_, 0,
               lb1 + (size_t)i * DFF_, nullptr, 0, 0, L_, DFF_, D_, 1.f, 1, 1);
        gemmNN(ff_b, DFF_, 0, lW2_b + (size_t)i * DFF_ * D_, D_, 0, latbuf, nullptr, D_, 0,
               lb2 + (size_t)i * D_, latbuf, D_, 0, L_, D_, DFF_, 1.f, 0, 1);
    }

    // ---- decoder attention ----
    layernorm_rows<<<S_, 256, 0, stream>>>(xbuf, decq, n2g, n2b, nullptr, dqin_b, D_);
    layernorm_rows<<<S_, 256, 0, stream>>>(xbuf, nullptr, n2g, n2b, nullptr, knb_b, D_);
    gemmNN(dqin_b, D_, 0, dWq_b, D_, 0, nullptr, dq_b, D_, 0, nullptr, nullptr, 0, 0, S_, D_, D_, 1.f, 0, 1);
    gemmNN(knb_b,  D_, 0, dWk_b, D_, 0, nullptr, dk_b, D_, 0, nullptr, nullptr, 0, 0, S_, D_, D_, 1.f, 0, 1);
    gemmNN(knb_b,  D_, 0, dWv_b, D_, 0, dvv, dv_b, D_, 0, nullptr, nullptr, 0, 0, S_, D_, D_, 1.f, 0, 1);
    transpose_bf16_kernel<<<dim3(D_ / 32, S_ / 32), 256, 0, stream>>>(dv_b, dvT_b, S_, D_);
    zero_bf16_kernel<<<1, 64, 0, stream>>>(dvT_b + (size_t)D_ * S_, 64);
    group_vsum_kernel<<<(L_ * D_ + 255) / 256, 256, 0, stream>>>(dvv, vgsb, L_ * D_);
    dec_flash_attn<<<dim3(S_ / 16, H_), 32, 0, stream>>>(dq_b, dk_b, dvT_b, dvv, latsc, vgsb, mixp, dao_b);
    gemmNN(dao_b, D_, 0, dWo_b, D_, 0, ybuf, nullptr, D_, 0, dbo, xbuf, D_, 0, S_, D_, D_, 1.f, 0, 1);

    // ---- final LN -> output ----
    layernorm_rows<<<S_, 256, 0, stream>>>(ybuf, nullptr, fing, finb, (float*)d_out, nullptr, D_);
}
